// RNN_MAL_26319559590679
// MI455X (gfx1250) — matrix operand order repaired and checked by reading, not compiled, not run on hardware
//
#include <hip/hip_runtime.h>
#include <hip/hip_bf16.h>
#include <stdint.h>

#define N_INST 2048
#define SEQ_LEN 100
#define EMB 128
#define HID 128
#define GATES 512     // 4*HID
#define TWO_H 256
#define KOUT 25600    // SEQ_LEN*TWO_H
#define NUM_BAGS 128
#define NUM_CLASSES 10
#define NKT_OUT 800   // KOUT/32

typedef __attribute__((ext_vector_type(16))) __bf16 v16bf;
typedef __attribute__((ext_vector_type(8)))  float  v8f;

union Frag16 { uint4 u[2]; unsigned short s[16]; v16bf v; };
union CFrag  { v8f v; float f[8]; };

__device__ __forceinline__ unsigned short f2bf(float f) {
  unsigned int u = __float_as_uint(f);
  unsigned int r = (u + 0x7FFFu + ((u >> 16) & 1u)) >> 16;
  return (unsigned short)r;
}
__device__ __forceinline__ float sigmoidf_(float x) {
  return 1.0f / (1.0f + __expf(-x));
}

// ---------------- prep kernels ----------------

__global__ void zero_kernel(float* p, int n) {
  int i = blockIdx.x * blockDim.x + threadIdx.x;
  if (i < n) p[i] = 0.0f;
}

__global__ void cvt_emb_kernel(const float* __restrict__ src,
                               unsigned short* __restrict__ dst, int n) {
  int i = blockIdx.x * blockDim.x + threadIdx.x;
  if (i < n) dst[i] = f2bf(src[i]);
}

// Pack [W;U] (K=256, N=512) per direction into WMMA B-fragment tiles:
// layout [dir][kt(8)][nt(32)][lane(32)][e(16)]
// lane l supplies B[k = kt*32 + (l>>4)*8 + (e&7) + (e>>3)*16][n = nt*16 + (l&15)]
__global__ void pack_wu_kernel(const float* __restrict__ Wf, const float* __restrict__ Uf,
                               const float* __restrict__ Wb, const float* __restrict__ Ub,
                               unsigned short* __restrict__ dst) {
  int tid = blockIdx.x * blockDim.x + threadIdx.x;
  if (tid >= 2 * 8 * 32 * 32 * 16) return;
  int e    = tid & 15;
  int lane = (tid >> 4) & 31;
  int nt   = (tid >> 9) & 31;
  int kt   = (tid >> 14) & 7;
  int dir  = tid >> 17;
  int k = kt * 32 + (lane >> 4) * 8 + (e & 7) + (e >> 3) * 16;     // 0..255
  int n = nt * 16 + (lane & 15);              // 0..511
  const float* W = dir ? Wb : Wf;
  const float* U = dir ? Ub : Uf;
  float v = (k < EMB) ? W[k * GATES + n] : U[(k - EMB) * GATES + n];
  dst[tid] = f2bf(v);
}

// Pack Wout [25600,10] into B-fragment tiles padded to N=16: [kt(800)][lane(32)][e(16)]
__global__ void pack_wout_kernel(const float* __restrict__ Wout,
                                 unsigned short* __restrict__ dst) {
  int tid = blockIdx.x * blockDim.x + threadIdx.x;
  if (tid >= NKT_OUT * 512) return;
  int e    = tid & 15;
  int lane = (tid >> 4) & 31;
  int kt   = tid >> 9;
  int k = kt * 32 + (lane >> 4) * 8 + (e & 7) + (e >> 3) * 16;     // < 25600
  int n = lane & 15;
  float v = (n < NUM_CLASSES) ? Wout[k * NUM_CLASSES + n] : 0.0f;
  dst[tid] = f2bf(v);
}

// ---------------- LSTM step: z = [xe_t | h] @ [W;U] + b, then cell update ----
// grid (128 row-tiles, 2 directions), block 256 (8 waves).
// Wave w owns gate-column tiles {w, w+8, w+16, w+24} -> i/f/g/o for hidden
// cols [16w,16w+16) land in matching lane/VGPR slots.
__global__ void lstm_step_kernel(const int* __restrict__ x_input,
                                 const unsigned short* __restrict__ emb_bf,
                                 const unsigned short* __restrict__ h_in,   // [2][N][HID]
                                 unsigned short* __restrict__ h_out,        // [2][N][HID]
                                 float* __restrict__ c_buf,                 // [2][N][HID]
                                 float* __restrict__ hcat,                  // [N][T][256]
                                 const unsigned short* __restrict__ wu_pack,
                                 const float* __restrict__ bias_f,
                                 const float* __restrict__ bias_b,
                                 int t) {
  const int d    = blockIdx.y;
  const int tt   = d ? (SEQ_LEN - 1 - t) : t;
  const int wave = threadIdx.x >> 5;
  const int lane = threadIdx.x & 31;
  const int m    = lane & 15;
  const int side = lane >> 4;
  const int grow = blockIdx.x * 16 + m;
  const float* bias = d ? bias_b : bias_f;

  v8f acc[4];
#pragma unroll
  for (int j = 0; j < 4; ++j) {
    float bv = bias[j * HID + wave * 16 + m];   // per-column bias, broadcast over rows
    v8f bvec = {bv, bv, bv, bv, bv, bv, bv, bv};
    acc[j] = bvec;
  }

  const int idx = x_input[grow * SEQ_LEN + tt];
  const unsigned short* erow = emb_bf + (size_t)idx * EMB;
  const unsigned short* hrow = h_in + ((size_t)d * N_INST + grow) * HID;

#pragma unroll
  for (int kt = 0; kt < 8; ++kt) {
    const unsigned short* rowp = (kt < 4) ? erow : hrow;
    const int kof = ((kt & 3) * 32) + side * 8;
    Frag16 a;
    a.u[0] = *(const uint4*)(rowp + kof);
    a.u[1] = *(const uint4*)(rowp + kof + 16);
#pragma unroll
    for (int j = 0; j < 4; ++j) {
      int tile = ((d * 8 + kt) * 32) + (j * 8 + wave);
      const unsigned short* bp = wu_pack + (size_t)tile * 512 + lane * 16;
      Frag16 b;
      b.u[0] = *(const uint4*)(bp);
      b.u[1] = *(const uint4*)(bp + 8);
      acc[j] = __builtin_amdgcn_wmma_f32_16x16x32_bf16(
          false, a.v, false, b.v, (short)0, acc[j], false, false);
    }
  }

  CFrag zi, zf, zg, zo;
  zi.v = acc[0]; zf.v = acc[1]; zg.v = acc[2]; zo.v = acc[3];
  const int col = wave * 16 + m;  // hidden unit (C-fragment column = lane&15)
#pragma unroll
  for (int r = 0; r < 8; ++r) {
    int row = blockIdx.x * 16 + r + side * 8;
    float iv = sigmoidf_(zi.f[r]);
    float fv = sigmoidf_(zf.f[r]);
    float gv = tanhf(zg.f[r]);
    float ov = sigmoidf_(zo.f[r]);
    size_t ci = ((size_t)d * N_INST + row) * HID + col;
    float cn = fv * c_buf[ci] + iv * gv;
    c_buf[ci] = cn;
    float hv = ov * tanhf(cn);
    h_out[ci] = f2bf(hv);
    hcat[((size_t)row * SEQ_LEN + tt) * TWO_H + d * HID + col] = hv;
  }
}

// ---------------- LayerNorm stats: one wave per (n,t) row of 256 ------------
__global__ void ln_stats_kernel(const float* __restrict__ hcat,
                                float* __restrict__ mu, float* __restrict__ rstd) {
  const int wave = threadIdx.x >> 5;
  const int lane = threadIdx.x & 31;
  const int row  = blockIdx.x * 8 + wave;       // 0..204799
  const float* p = hcat + (size_t)row * TWO_H + lane * 8;
  float4 a = *(const float4*)p;
  float4 b = *(const float4*)(p + 4);
  float s = a.x + a.y + a.z + a.w + b.x + b.y + b.z + b.w;
  float q = a.x*a.x + a.y*a.y + a.z*a.z + a.w*a.w +
            b.x*b.x + b.y*b.y + b.z*b.z + b.w*b.w;
#pragma unroll
  for (int o = 16; o > 0; o >>= 1) {
    s += __shfl_xor(s, o, 32);
    q += __shfl_xor(q, o, 32);
  }
  if (lane == 0) {
    float mean = s * (1.0f / TWO_H);
    float var  = q * (1.0f / TWO_H) - mean * mean;
    mu[row]   = mean;
    rstd[row] = rsqrtf(var + 1e-6f);
  }
}

// ---------------- dense: logits[n,c] = LN(flat[n]) @ Wout -------------------
// LN applied on the fly via mu/rstd + gamma/beta; WMMA over K=25600.
__global__ void logits_kernel(const float* __restrict__ hcat,
                              const float* __restrict__ mu,
                              const float* __restrict__ rstd,
                              const float* __restrict__ gamma,
                              const float* __restrict__ beta,
                              const unsigned short* __restrict__ wout_pack,
                              float* __restrict__ logits) {
  __shared__ float lds[256];
  const int tid  = threadIdx.x;
  const int wave = tid >> 5;
  const int lane = tid & 31;
  const int m    = lane & 15;
  const int side = lane >> 4;
  const int grow = blockIdx.x * 16 + m;
  lds[tid] = 0.0f;
  __syncthreads();

  v8f acc = {0, 0, 0, 0, 0, 0, 0, 0};
  for (int kt = wave; kt < NKT_OUT; kt += 8) {     // uniform per wave
    const int t  = kt >> 3;                         // 256-wide LN segment
    const int c0 = (kt & 7) * 32 + side * 8;        // col within segment
    const float muv = mu[grow * SEQ_LEN + t];
    const float rs  = rstd[grow * SEQ_LEN + t];
    const float* p = hcat + (size_t)grow * KOUT + kt * 32 + side * 8;
    float4 f0 = *(const float4*)p;
    float4 f1 = *(const float4*)(p + 4);
    float4 f2 = *(const float4*)(p + 16);
    float4 f3 = *(const float4*)(p + 20);
    float v0[8] = {f0.x, f0.y, f0.z, f0.w, f1.x, f1.y, f1.z, f1.w};
    float v1[8] = {f2.x, f2.y, f2.z, f2.w, f3.x, f3.y, f3.z, f3.w};
    Frag16 a;
#pragma unroll
    for (int e = 0; e < 8; ++e) {
      int ca = c0 + e;
      int cb = c0 + 16 + e;
      a.s[e]     = f2bf((v0[e] - muv) * rs * gamma[ca] + beta[ca]);
      a.s[e + 8] = f2bf((v1[e] - muv) * rs * gamma[cb] + beta[cb]);
    }
    const unsigned short* bp = wout_pack + (size_t)kt * 512 + lane * 16;
    Frag16 b;
    b.u[0] = *(const uint4*)(bp);
    b.u[1] = *(const uint4*)(bp + 8);
    acc = __builtin_amdgcn_wmma_f32_16x16x32_bf16(
        false, a.v, false, b.v, (short)0, acc, false, false);
  }

  CFrag cf; cf.v = acc;
#pragma unroll
  for (int r = 0; r < 8; ++r) {
    atomicAdd(&lds[(r + side * 8) * 16 + m], cf.f[r]);  // cross-wave K reduction
  }
  __syncthreads();
  if (tid < 16 * NUM_CLASSES) {
    int rr = tid / NUM_CLASSES, cc = tid % NUM_CLASSES;
    logits[(blockIdx.x * 16 + rr) * NUM_CLASSES + cc] = lds[rr * 16 + cc];
  }
}

// ---------------- bag pooling ----------------
__global__ void pool_kernel(const float* __restrict__ logits,
                            const int* __restrict__ spans,
                            const float* __restrict__ bout,
                            float* __restrict__ out) {
  int bag = blockIdx.x;
  int c   = threadIdx.x;
  if (c >= NUM_CLASSES) return;
  int s0 = spans[bag * 2 + 0];
  int s1 = spans[bag * 2 + 1];
  float s = bout[c];
  for (int n = s0; n < s1; ++n) s += logits[n * NUM_CLASSES + c];
  out[bag * NUM_CLASSES + c] = s;
}

// ---------------- host orchestration ----------------
extern "C" void kernel_launch(void* const* d_in, const int* in_sizes, int n_in,
                              void* d_out, int out_size, void* d_ws, size_t ws_size,
                              hipStream_t stream) {
  (void)in_sizes; (void)n_in; (void)out_size; (void)ws_size;
  const int*   x_input = (const int*)d_in[0];
  const int*   spans   = (const int*)d_in[1];
  const float* emb     = (const float*)d_in[3];
  const float* Wf      = (const float*)d_in[4];
  const float* Uf      = (const float*)d_in[5];
  const float* bf      = (const float*)d_in[6];
  const float* Wb      = (const float*)d_in[7];
  const float* Ub      = (const float*)d_in[8];
  const float* bb      = (const float*)d_in[9];
  const float* gamma   = (const float*)d_in[10];
  const float* beta    = (const float*)d_in[11];
  const float* Wout    = (const float*)d_in[12];
  const float* bout    = (const float*)d_in[13];
  float* out = (float*)d_out;

  char* ws = (char*)d_ws;
  size_t off = 0;
  auto alloc = [&](size_t bytes) -> char* {
    char* p = ws + off;
    off += (bytes + 255) & ~(size_t)255;
    return p;
  };
  unsigned short* emb_bf  = (unsigned short*)alloc((size_t)10000 * EMB * 2);
  unsigned short* wu_pack = (unsigned short*)alloc((size_t)2 * 8 * 32 * 32 * 16 * 2);
  unsigned short* wout_pk = (unsigned short*)alloc((size_t)NKT_OUT * 512 * 2);
  unsigned short* hbuf0   = (unsigned short*)alloc((size_t)2 * N_INST * HID * 2);
  unsigned short* hbuf1   = (unsigned short*)alloc((size_t)2 * N_INST * HID * 2);
  float* c_buf  = (float*)alloc((size_t)2 * N_INST * HID * 4);
  float* hcat   = (float*)alloc((size_t)N_INST * SEQ_LEN * TWO_H * 4);
  float* mu     = (float*)alloc((size_t)N_INST * SEQ_LEN * 4);
  float* rstd   = (float*)alloc((size_t)N_INST * SEQ_LEN * 4);
  float* logits = (float*)alloc((size_t)N_INST * NUM_CLASSES * 4);

  // h(t=0)=0, c=0
  {
    int nh = (2 * N_INST * HID * 2) / 4;   // hbuf0 viewed as floats
    zero_kernel<<<(nh + 255) / 256, 256, 0, stream>>>((float*)hbuf0, nh);
    int nc = 2 * N_INST * HID;
    zero_kernel<<<(nc + 255) / 256, 256, 0, stream>>>(c_buf, nc);
  }
  cvt_emb_kernel<<<(10000 * EMB + 255) / 256, 256, 0, stream>>>(emb, emb_bf, 10000 * EMB);
  pack_wu_kernel<<<(2 * 8 * 32 * 32 * 16) / 256, 256, 0, stream>>>(Wf, Uf, Wb, Ub, wu_pack);
  pack_wout_kernel<<<(NKT_OUT * 512) / 256, 256, 0, stream>>>(Wout, wout_pk);

  for (int t = 0; t < SEQ_LEN; ++t) {
    const unsigned short* hin = (t & 1) ? hbuf1 : hbuf0;
    unsigned short* hout      = (t & 1) ? hbuf0 : hbuf1;
    lstm_step_kernel<<<dim3(N_INST / 16, 2), 256, 0, stream>>>(
        x_input, emb_bf, hin, hout, c_buf, hcat, wu_pack, bf, bb, t);
  }

  ln_stats_kernel<<<(N_INST * SEQ_LEN) / 8, 256, 0, stream>>>(hcat, mu, rstd);
  logits_kernel<<<N_INST / 16, 256, 0, stream>>>(hcat, mu, rstd, gamma, beta, wout_pk, logits);
  pool_kernel<<<NUM_BAGS, 32, 0, stream>>>(logits, spans, bout, out);
}
